// FraudGNN_91018946937012
// MI455X (gfx1250) — matrix operand order repaired and checked by reading, not compiled, not run on hardware
//
#include <hip/hip_runtime.h>

// ---------------------------------------------------------------------------
// GraphSAGE (3 layers) + classifier for MI455X (gfx1250, wave32).
// Strategy: transform-then-aggregate (linearity of segment_sum), bf16 WMMA
// GEMMs with f32 accumulation, f32 atomic scatter-add (L2-resident working
// set), bf16 activation storage between layers.
// Workspace use: ~116 MB (xb 25.6M + wT 32K + G 51.2M + agg 25.6M +
//                hbf 12.8M + deg 0.4M).
// ---------------------------------------------------------------------------

#define NN 100000
#define NE 1600000
#define DIN 128
#define DH 64

typedef __attribute__((ext_vector_type(16))) __bf16 v16bf;
typedef __attribute__((ext_vector_type(8)))  float  v8f;
typedef __attribute__((ext_vector_type(8)))  unsigned short ushort8;
typedef __attribute__((ext_vector_type(16))) unsigned short ushort16;

union Frag { ushort8 h[2]; ushort16 w; };

__device__ __forceinline__ unsigned short f32_bf16(float f) {
  unsigned int u = __builtin_bit_cast(unsigned int, f);
  u += 0x7FFFu + ((u >> 16) & 1u);          // round-to-nearest-even
  return (unsigned short)(u >> 16);
}
__device__ __forceinline__ float bf16_f32(unsigned short h) {
  unsigned int u = ((unsigned int)h) << 16;
  return __builtin_bit_cast(float, u);
}

// C[N x 128](f32) = A[N x K](bf16) @ B[K x 128](bf16),
// B supplied transposed: Wt[m][k], m = output column (0..63 = Wl, 64..127 = Wr).
// One wave per 16-row tile, 8 column tiles of 16; weights staged in padded LDS.
template <int K>
__global__ void __launch_bounds__(256)
gemm_wmma(const unsigned short* __restrict__ A,
          const unsigned short* __restrict__ Wt,
          float* __restrict__ C, int nRows) {
  constexpr int KP = K + 8;                 // pad: 16B rows land on distinct banks
  constexpr int KS = K / 32;                // WMMA K-steps (bf16: K=32 each)
  __shared__ unsigned short ldsW[128 * KP];

  for (int idx = threadIdx.x; idx < 128 * K; idx += 256) {
    int m = idx / K, k = idx % K;
    ldsW[m * KP + k] = Wt[idx];
  }
  __syncthreads();

  const int lane = threadIdx.x & 31;
  const int wave = threadIdx.x >> 5;
  const int rowBase = (blockIdx.x * 8 + wave) * 16;
  if (rowBase >= nRows) return;
  const int half = lane >> 4;               // lane-half selects K sub-block
  const int l16  = lane & 15;

  // A fragment, ISA 16-bit 16x32 layout: lanes 0-15 rows M, elems 0..7 = K
  // (half*8)+0..7, elems 8..15 = K 16+half*8 .. +7.
  Frag af[KS];
  const unsigned short* Arow = A + (size_t)(rowBase + l16) * K;
#pragma unroll
  for (int s = 0; s < KS; ++s) {
    af[s].h[0] = *(const ushort8*)(Arow + s * 32 + half * 8);
    af[s].h[1] = *(const ushort8*)(Arow + s * 32 + 16 + half * 8);
  }

#pragma unroll
  for (int ct = 0; ct < 8; ++ct) {
    v8f acc = {0.f, 0.f, 0.f, 0.f, 0.f, 0.f, 0.f, 0.f};
    // B fragment 32x16: col = ct*16 + l16; elems j<8 -> K = k0 + half*8 + j, j>=8 -> K = k0 + 16 + half*8 + (j-8)
    const unsigned short* wp = ldsW + (ct * 16 + l16) * KP + half * 8;
#pragma unroll
    for (int s = 0; s < KS; ++s) {
      Frag bfr;
      bfr.h[0] = *(const ushort8*)(wp + s * 32);
      bfr.h[1] = *(const ushort8*)(wp + s * 32 + 16);
      acc = __builtin_amdgcn_wmma_f32_16x16x32_bf16(
          false, __builtin_bit_cast(v16bf, af[s].w),
          false, __builtin_bit_cast(v16bf, bfr.w),
          (short)0, acc, false, false);
    }
    // C layout: lane<16 -> M = v, N = lane; lane>=16 -> M = 8+v, N = lane-16
    float* crow = C + (size_t)(rowBase + half * 8) * 128 + ct * 16 + l16;
#pragma unroll
    for (int v = 0; v < 8; ++v) crow[(size_t)v * 128] = acc[v];
  }
}

__global__ void cvt_bf16(const float* __restrict__ in,
                         unsigned short* __restrict__ out, int n) {
  int i = blockIdx.x * 256 + threadIdx.x;
  if (i < n) out[i] = f32_bf16(in[i]);
}

__global__ void deg_count(const long long* __restrict__ dst,
                          float* __restrict__ deg, int nE) {
  int e = blockIdx.x * 256 + threadIdx.x;
  if (e < nE)
    __hip_atomic_fetch_add(&deg[dst[e]], 1.0f, __ATOMIC_RELAXED,
                           __HIP_MEMORY_SCOPE_AGENT);
}

__global__ void deg_invert(float* __restrict__ deg, int n) {
  int i = blockIdx.x * 256 + threadIdx.x;
  if (i < n) deg[i] = 1.0f / fmaxf(deg[i], 1.0f);
}

// agg[dst] += y[src]; y = columns 0..63 of G (stride 128)
__global__ void scatter_add(const long long* __restrict__ src,
                            const long long* __restrict__ dst,
                            const float* __restrict__ G,
                            float* __restrict__ agg, int nE) {
  int t = blockIdx.x * 256 + threadIdx.x;
  int e = t >> 6, f = t & 63;
  if (e < nE) {
    float v = G[(size_t)src[e] * 128 + f];
    __hip_atomic_fetch_add(&agg[(size_t)dst[e] * 64 + f], v, __ATOMIC_RELAXED,
                           __HIP_MEMORY_SCOPE_AGENT);
  }
}

// h = relu(agg * deg_inv + bl + self), self = columns 64..127 of G; store bf16
__global__ void finalize_layer(const float* __restrict__ agg,
                               const float* __restrict__ G,
                               const float* __restrict__ bl,
                               const float* __restrict__ dinv,
                               unsigned short* __restrict__ hbf, int n) {
  int t = blockIdx.x * 256 + threadIdx.x;
  int node = t >> 6, f = t & 63;
  if (node < n) {
    float v = agg[(size_t)node * 64 + f] * dinv[node] + bl[f] +
              G[(size_t)node * 128 + 64 + f];
    hbf[(size_t)node * 64 + f] = f32_bf16(fmaxf(v, 0.0f));
  }
}

__global__ void classify(const unsigned short* __restrict__ h,
                         const float* __restrict__ Wc,
                         const float* __restrict__ bc,
                         float* __restrict__ out, int n) {
  int node = blockIdx.x * 256 + threadIdx.x;
  if (node < n) {
    float s = bc[0];
#pragma unroll
    for (int f = 0; f < 64; ++f)
      s += bf16_f32(h[(size_t)node * 64 + f]) * Wc[f];
    out[node] = s;
  }
}

extern "C" void kernel_launch(void* const* d_in, const int* in_sizes, int n_in,
                              void* d_out, int out_size, void* d_ws,
                              size_t ws_size, hipStream_t stream) {
  (void)in_sizes; (void)n_in; (void)out_size; (void)ws_size;
  const float*     x  = (const float*)d_in[0];
  const long long* ei = (const long long*)d_in[1];          // int64 [2, E]
  const float* Wl[3] = {(const float*)d_in[2], (const float*)d_in[5],
                        (const float*)d_in[8]};
  const float* bl[3] = {(const float*)d_in[3], (const float*)d_in[6],
                        (const float*)d_in[9]};
  const float* Wr[3] = {(const float*)d_in[4], (const float*)d_in[7],
                        (const float*)d_in[10]};
  const float* Wc = (const float*)d_in[11];
  const float* bc = (const float*)d_in[12];
  const long long* src = ei;
  const long long* dst = ei + NE;

  char* ws = (char*)d_ws;
  size_t off = 0;
  auto carve = [&](size_t bytes) {
    char* p = ws + off;
    off = (off + bytes + 255) & ~(size_t)255;
    return p;
  };
  unsigned short* xb  = (unsigned short*)carve((size_t)NN * DIN * 2);
  unsigned short* wT  = (unsigned short*)carve(128 * 128 * 2);
  float*          G   = (float*)carve((size_t)NN * 128 * 4);
  float*          agg = (float*)carve((size_t)NN * 64 * 4);
  unsigned short* hbf = (unsigned short*)carve((size_t)NN * 64 * 2);
  float*          dnv = (float*)carve((size_t)NN * 4);

  // degrees -> 1/max(deg,1)
  hipMemsetAsync(dnv, 0, (size_t)NN * 4, stream);
  deg_count<<<(NE + 255) / 256, 256, 0, stream>>>(dst, dnv, NE);
  deg_invert<<<(NN + 255) / 256, 256, 0, stream>>>(dnv, NN);

  // x -> bf16 once
  cvt_bf16<<<(NN * DIN + 255) / 256, 256, 0, stream>>>(x, xb, NN * DIN);

  const int rowBlocks = (NN / 16 + 7) / 8;  // 782 blocks x 8 waves
  for (int l = 0; l < 3; ++l) {
    const int K = (l == 0) ? DIN : DH;
    // pack [Wl; Wr] transposed-layout bf16 weights (rows = output columns)
    cvt_bf16<<<(64 * K + 255) / 256, 256, 0, stream>>>(Wl[l], wT, 64 * K);
    cvt_bf16<<<(64 * K + 255) / 256, 256, 0, stream>>>(Wr[l], wT + 64 * K,
                                                       64 * K);
    const unsigned short* Ain = (l == 0) ? xb : hbf;
    if (K == 128)
      gemm_wmma<128><<<rowBlocks, 256, 0, stream>>>(Ain, wT, G, NN);
    else
      gemm_wmma<64><<<rowBlocks, 256, 0, stream>>>(Ain, wT, G, NN);

    hipMemsetAsync(agg, 0, (size_t)NN * 64 * 4, stream);
    scatter_add<<<(int)(((size_t)NE * 64 + 255) / 256), 256, 0, stream>>>(
        src, dst, G, agg, NE);
    finalize_layer<<<(NN * 64 + 255) / 256, 256, 0, stream>>>(agg, G, bl[l],
                                                              dnv, hbf, NN);
  }
  classify<<<(NN + 255) / 256, 256, 0, stream>>>(hbf, Wc, bc, (float*)d_out,
                                                 NN);
}
